// Attention_87119116632809
// MI455X (gfx1250) — matrix operand order repaired and checked by reading, not compiled, not run on hardware
//
#include <hip/hip_runtime.h>
#include <hip/hip_bf16.h>

// ---------------------------------------------------------------------------
// Shapes (fixed by the reference)
// ---------------------------------------------------------------------------
#define BATCH      16
#define SEQ        1024          // N = 32*32
#define HEADS      8
#define DHEAD      64
#define INNER      512           // HEADS*DHEAD
#define INP        512
#define OUP        512
#define QKV_COLS   1536          // 3*INNER
#define SCALE_Q    0.125f        // DHEAD^-0.5

typedef __attribute__((ext_vector_type(16))) __bf16 v16bf;
typedef __attribute__((ext_vector_type(8)))  __bf16 v8bf;
typedef __attribute__((ext_vector_type(8)))  float  v8f;

// ---------------------------------------------------------------------------
// WMMA fragment helpers (contiguous k = 16*(lane>>4) + e in A and B alike, wave32)
// ---------------------------------------------------------------------------
// A fragment, 16x32 bf16, source row-major with leading dim `ld`:
//   lane L: row = L&15 ; K = kb..kb+15, kb = (L>>4)*16
__device__ __forceinline__ v16bf load_a_frag(const __bf16* src, int ld) {
    const int lane = threadIdx.x & 31;
    const int r    = lane & 15;
    const int kb   = (lane >> 4) << 4;
    const __bf16* p = src + (size_t)r * ld;
    v8bf lo = *reinterpret_cast<const v8bf*>(p + kb);
    v8bf hi = *reinterpret_cast<const v8bf*>(p + kb + 8);
    v16bf a;
#pragma unroll
    for (int i = 0; i < 8; ++i) { a[i] = lo[i]; a[i + 8] = hi[i]; }
    return a;
}

// B fragment, 32x16 bf16. srcT holds B^T row-major: srcT[c*ld + k] = B[k][c].
//   lane L: col = L&15 ; K = (L>>4)*16 + e  -> 32 contiguous bytes per lane.
__device__ __forceinline__ v16bf load_b_frag(const __bf16* srcT, int ld) {
    const int lane = threadIdx.x & 31;
    const int c    = lane & 15;
    const int koff = (lane >> 4) << 4;
    return *reinterpret_cast<const v16bf*>(srcT + (size_t)c * ld + koff);
}

__device__ __forceinline__ v8f wmma_bf16(v16bf a, v16bf b, v8f c) {
    return __builtin_amdgcn_wmma_f32_16x16x32_bf16(
        /*neg_a=*/false, a, /*neg_b=*/false, b,
        /*c_mod=*/(short)0, c, /*reuse_a=*/false, /*reuse_b=*/false);
}

// ---------------------------------------------------------------------------
// Prep kernels
// ---------------------------------------------------------------------------
__global__ __launch_bounds__(256) void cvt_to_bf16(const float* __restrict__ src,
                                                   __bf16* __restrict__ dst, int n) {
    int i = blockIdx.x * 256 + threadIdx.x;
    if (i < n) dst[i] = (__bf16)src[i];
}

// dst[c*R + r] = src[r*C + c]  (transpose + convert -> B^T layout)
__global__ __launch_bounds__(256) void transpose_cvt(const float* __restrict__ src,
                                                     __bf16* __restrict__ dst,
                                                     int R, int C) {
    int i = blockIdx.x * 256 + threadIdx.x;
    if (i < R * C) {
        int r = i / C, c = i - r * C;
        dst[(size_t)c * R + r] = (__bf16)src[i];
    }
}

// Dense relative-position bias, pre-permuted into WMMA C-fragment order:
//   biasB[ ((h*64 + it)*32 + mt)*512 + lane*16 + e ]
//     e = fr*8 + g ; source (i,j) = (it*16 + g + 8*(lane>>4), mt*32 + fr*16 + (lane&15))
// so each lane in the attention hot loop does ONE contiguous 32B load.
__global__ __launch_bounds__(256) void build_bias(const float* __restrict__ table,
                                                  const int* __restrict__ ridx,
                                                  __bf16* __restrict__ bias) {
    int t = blockIdx.x * 256 + threadIdx.x;          // 8 * 1024 * 1024
    int e    = t & 15;
    int lane = (t >> 4) & 31;
    int mt   = (t >> 9) & 31;
    int it   = (t >> 14) & 63;
    int h    = t >> 20;
    int g  = e & 7, fr = e >> 3;
    int i = it * 16 + g + 8 * (lane >> 4);
    int j = mt * 32 + fr * 16 + (lane & 15);
    bias[t] = (__bf16)table[(size_t)ridx[i * SEQ + j] * HEADS + h];
}

// ---------------------------------------------------------------------------
// Kernel 1: QKV projection  (x[16384,512] @ W_qkv[512,1536]).
// Fully-unrolled K loop + explicit 2-deep fragment pipeline (i&1 indexing is
// constant after unroll -> SSA, no copies; chunk i+1 loads issue before chunk
// i WMMAs -> partial s_wait_loadcnt, load/WMMA overlap).
// one wave -> 16x64 output tile ; scatter to q / k / v head layouts
// ---------------------------------------------------------------------------
__global__ __launch_bounds__(256) void qkv_gemm(const __bf16* __restrict__ xB,
                                                const __bf16* __restrict__ WqkvT,
                                                __bf16* __restrict__ qB,
                                                __bf16* __restrict__ kB,
                                                __bf16* __restrict__ vB) {
    const int wid  = blockIdx.x * 8 + (threadIdx.x >> 5);   // 24576 waves
    const int cg   = wid % 24;                              // 64-col group of 1536
    const int rt   = (wid / 24) & 63;                       // 16-row tile of 1024
    const int b    = wid / (24 * 64);
    const int lane = threadIdx.x & 31;

    const __bf16* abase = xB + ((size_t)(b * SEQ + rt * 16)) * INP;
    const __bf16* wbase = WqkvT + ((size_t)(cg * 64)) * INP;

    v8f acc[4];
#pragma unroll
    for (int j = 0; j < 4; ++j)
#pragma unroll
        for (int g = 0; g < 8; ++g) acc[j][g] = 0.f;

    v16bf aP[2], bP[2][4];
    aP[0] = load_a_frag(abase, INP);
#pragma unroll
    for (int j = 0; j < 4; ++j) bP[0][j] = load_b_frag(wbase + (size_t)(j * 16) * INP, INP);

#pragma unroll
    for (int i = 0; i < INP / 32; ++i) {
        const int cur = i & 1, nxt = cur ^ 1;
        if (i + 1 < INP / 32) {                      // prefetch chunk i+1
            aP[nxt] = load_a_frag(abase + (i + 1) * 32, INP);
#pragma unroll
            for (int j = 0; j < 4; ++j)
                bP[nxt][j] = load_b_frag(wbase + (size_t)(j * 16) * INP + (i + 1) * 32, INP);
        }
#pragma unroll
        for (int j = 0; j < 4; ++j) acc[j] = wmma_bf16(aP[cur], bP[cur][j], acc[j]);
    }

    const int colL = lane & 15, half = lane >> 4;
#pragma unroll
    for (int j = 0; j < 4; ++j) {
#pragma unroll
        for (int g = 0; g < 8; ++g) {
            int row = rt * 16 + g + 8 * half;
            int col = cg * 64 + j * 16 + colL;
            float v = acc[j][g];
            if (col < INNER) {                       // Q (pre-scaled)
                int h = col >> 6, d = col & 63;
                qB[(((size_t)(b * HEADS + h)) * SEQ + row) * DHEAD + d] =
                    (__bf16)(v * SCALE_Q);
            } else if (col < 2 * INNER) {            // K
                int cc = col - INNER, h = cc >> 6, d = cc & 63;
                kB[(((size_t)(b * HEADS + h)) * SEQ + row) * DHEAD + d] = (__bf16)v;
            } else {                                 // V (coalesced; transposed later)
                int cc = col - 2 * INNER, h = cc >> 6, d = cc & 63;
                vB[(((size_t)(b * HEADS + h)) * SEQ + row) * DHEAD + d] = (__bf16)v;
            }
        }
    }
}

// ---------------------------------------------------------------------------
// Kernel 1b: LDS-tiled transpose vB[bh][n][64] -> vtB[bh][64][n]
// coalesced 16B vector accesses on both sides (avoids 2B scattered stores)
// ---------------------------------------------------------------------------
__global__ __launch_bounds__(256) void transpose_v(const __bf16* __restrict__ vB,
                                                   __bf16* __restrict__ vtB) {
    __shared__ __bf16 tile[64][72];                  // 72: bank-conflict padding
    const int bh = blockIdx.x >> 4;
    const int n0 = (blockIdx.x & 15) * 64;
    const __bf16* src = vB + ((size_t)bh * SEQ + n0) * DHEAD;
#pragma unroll
    for (int it = 0; it < 2; ++it) {
        int idx = it * 2048 + threadIdx.x * 8;
        int r = idx >> 6, c = idx & 63;
        v8bf v = *reinterpret_cast<const v8bf*>(src + (size_t)r * DHEAD + c);
#pragma unroll
        for (int e = 0; e < 8; ++e) tile[r][c + e] = v[e];
    }
    __syncthreads();
#pragma unroll
    for (int it = 0; it < 2; ++it) {
        int idx = it * 2048 + threadIdx.x * 8;
        int d = idx >> 6, nn = idx & 63;
        v8bf o;
#pragma unroll
        for (int e = 0; e < 8; ++e) o[e] = tile[nn + e][d];
        *reinterpret_cast<v8bf*>(vtB + ((size_t)bh * DHEAD + d) * SEQ + n0 + nn) = o;
    }
}

// ---------------------------------------------------------------------------
// Kernel 2: flash attention. one wave = 16 query rows; 32 keys / step,
// two steps per loop trip with ping-pong K-fragment buffers (no copies).
// block = 8 waves on the same (b,h) -> K/V/bias tiles hit L0/L2.
// ---------------------------------------------------------------------------
__global__ __launch_bounds__(256) void attn_kernel(const __bf16* __restrict__ qB,
                                                   const __bf16* __restrict__ kB,
                                                   const __bf16* __restrict__ vtB,
                                                   const __bf16* __restrict__ biasB,
                                                   __bf16* __restrict__ aoB) {
    __shared__ __align__(16) __bf16 plds[8][16 * 32];   // per-wave P tile bounce

    const int warp = threadIdx.x >> 5;
    const int lane = threadIdx.x & 31;
    const int bh   = blockIdx.x >> 3;                   // 0..127
    const int rt   = (blockIdx.x & 7) * 8 + warp;       // 0..63
    const int b    = bh >> 3, h = bh & 7;
    const int colL = lane & 15, half = lane >> 4;

    const __bf16* qb = qB  + ((size_t)bh * SEQ + rt * 16) * DHEAD;
    const __bf16* kb = kB  + (size_t)bh * SEQ * DHEAD;
    const __bf16* vt = vtB + (size_t)bh * DHEAD * SEQ;
    const __bf16* biasTile = biasB + (((size_t)h * 64 + rt) * 32) * 512 + (size_t)lane * 16;

    // Q resident in registers: two 16x32 A fragments (d = 0..31, 32..63)
    v16bf qa0 = load_a_frag(qb, DHEAD);
    v16bf qa1 = load_a_frag(qb + 32, DHEAD);

    float mrow[8], lrow[8];
    v8f acc[4];
#pragma unroll
    for (int g = 0; g < 8; ++g) { mrow[g] = -__builtin_inff(); lrow[g] = 0.f; }
#pragma unroll
    for (int j = 0; j < 4; ++j)
#pragma unroll
        for (int g = 0; g < 8; ++g) acc[j][g] = 0.f;

    __bf16* pl = &plds[warp][0];

    // One 32-key step: consume K fragments kf[], prefetch K for key block mn
    // into nk[], do S WMMAs + bias + online softmax + LDS bounce + PV WMMAs.
    auto attn_step = [&](int m0, int mn, v16bf* kf, v16bf* nk) {
        // early independent loads (overlap the S WMMAs below)
        v16bf vf[4];
#pragma unroll
        for (int j = 0; j < 4; ++j)
            vf[j] = load_b_frag(vt + (size_t)(j * 16) * SEQ + m0, SEQ);
        v16bf bias16 = *reinterpret_cast<const v16bf*>(biasTile + (size_t)(m0 >> 5) * 512);
        nk[0] = load_b_frag(kb + (size_t)mn * DHEAD, DHEAD);
        nk[1] = load_b_frag(kb + (size_t)mn * DHEAD + 32, DHEAD);
        nk[2] = load_b_frag(kb + (size_t)(mn + 16) * DHEAD, DHEAD);
        nk[3] = load_b_frag(kb + (size_t)(mn + 16) * DHEAD + 32, DHEAD);

        // ---- S = (Q*scale) @ K^T : two 16x16 f32 tiles, K-depth 64 ----
        v8f s0, s1;
#pragma unroll
        for (int g = 0; g < 8; ++g) { s0[g] = 0.f; s1[g] = 0.f; }
        s0 = wmma_bf16(qa0, kf[0], s0);
        s0 = wmma_bf16(qa1, kf[1], s0);
        s1 = wmma_bf16(qa0, kf[2], s1);
        s1 = wmma_bf16(qa1, kf[3], s1);

        // ---- bias + online softmax (row lives in 16 lanes of one VGPR) ----
        float p0[8], p1[8], alpha[8];
#pragma unroll
        for (int g = 0; g < 8; ++g) {
            float t0 = s0[g] + (float)bias16[g];
            float t1 = s1[g] + (float)bias16[8 + g];
            float t = fmaxf(t0, t1);
            t = fmaxf(t, __shfl_xor(t, 1, 32));
            t = fmaxf(t, __shfl_xor(t, 2, 32));
            t = fmaxf(t, __shfl_xor(t, 4, 32));
            t = fmaxf(t, __shfl_xor(t, 8, 32));
            float mnew = fmaxf(mrow[g], t);
            alpha[g] = __expf(mrow[g] - mnew);
            p0[g] = __expf(t0 - mnew);
            p1[g] = __expf(t1 - mnew);
            float rs = p0[g] + p1[g];
            rs += __shfl_xor(rs, 1, 32);
            rs += __shfl_xor(rs, 2, 32);
            rs += __shfl_xor(rs, 4, 32);
            rs += __shfl_xor(rs, 8, 32);
            lrow[g] = lrow[g] * alpha[g] + rs;
            mrow[g] = mnew;
        }
#pragma unroll
        for (int j = 0; j < 4; ++j)
#pragma unroll
            for (int g = 0; g < 8; ++g) acc[j][g] *= alpha[g];

        // ---- C-layout -> A-layout through per-wave LDS tile ----
        // (DS ops are in-order within a wave, so stage N+1 stores cannot pass
        //  stage N loads; the explicit wait orders store->load.)
#pragma unroll
        for (int g = 0; g < 8; ++g) {
            int row = g + 8 * half;
            pl[row * 32 + colL]      = (__bf16)p0[g];
            pl[row * 32 + 16 + colL] = (__bf16)p1[g];
        }
        asm volatile("s_wait_dscnt 0x0" ::: "memory");
        v16bf pa = load_a_frag(pl, 32);

        // ---- acc += P @ V  (V^T rows are contiguous B fragments) ----
#pragma unroll
        for (int j = 0; j < 4; ++j) acc[j] = wmma_bf16(pa, vf[j], acc[j]);
    };

    // preload K fragments for m0 = 0, then ping-pong two steps per trip
    v16bf kfA[4], kfB[4];
    kfA[0] = load_b_frag(kb, DHEAD);
    kfA[1] = load_b_frag(kb + 32, DHEAD);
    kfA[2] = load_b_frag(kb + (size_t)16 * DHEAD, DHEAD);
    kfA[3] = load_b_frag(kb + (size_t)16 * DHEAD + 32, DHEAD);

    for (int m0 = 0; m0 < SEQ; m0 += 64) {
        attn_step(m0, m0 + 32, kfA, kfB);
        const int mn = (m0 + 64 < SEQ) ? m0 + 64 : 0;   // wrap: dead on last trip
        attn_step(m0 + 32, mn, kfB, kfA);
    }

    // ---- epilogue: normalize and store to [b*n, h*64+d] bf16 ----
    float inv[8];
#pragma unroll
    for (int g = 0; g < 8; ++g) inv[g] = 1.f / lrow[g];
#pragma unroll
    for (int j = 0; j < 4; ++j) {
#pragma unroll
        for (int g = 0; g < 8; ++g) {
            int row = b * SEQ + rt * 16 + g + 8 * half;
            int col = h * DHEAD + j * 16 + colL;
            aoB[(size_t)row * INNER + col] = (__bf16)(acc[j][g] * inv[g]);
        }
    }
}

// ---------------------------------------------------------------------------
// Kernel 3: output projection (ao[16384,512] @ W_out[512,512]) + bias, f32 out
// same fully-unrolled 2-deep pipeline as qkv_gemm
// ---------------------------------------------------------------------------
__global__ __launch_bounds__(256) void out_gemm(const __bf16* __restrict__ aoB,
                                                const __bf16* __restrict__ WoutT,
                                                const float* __restrict__ b_out,
                                                float* __restrict__ out) {
    const int wid  = blockIdx.x * 8 + (threadIdx.x >> 5);  // 8192 waves
    const int cg   = wid & 7;
    const int rt   = wid >> 3;                             // 0..1023
    const int lane = threadIdx.x & 31;

    const __bf16* abase = aoB + (size_t)rt * 16 * INNER;
    const __bf16* wbase = WoutT + ((size_t)(cg * 64)) * INNER;

    v8f acc[4];
#pragma unroll
    for (int j = 0; j < 4; ++j)
#pragma unroll
        for (int g = 0; g < 8; ++g) acc[j][g] = 0.f;

    v16bf aP[2], bP[2][4];
    aP[0] = load_a_frag(abase, INNER);
#pragma unroll
    for (int j = 0; j < 4; ++j) bP[0][j] = load_b_frag(wbase + (size_t)(j * 16) * INNER, INNER);

#pragma unroll
    for (int i = 0; i < INNER / 32; ++i) {
        const int cur = i & 1, nxt = cur ^ 1;
        if (i + 1 < INNER / 32) {                    // prefetch chunk i+1
            aP[nxt] = load_a_frag(abase + (i + 1) * 32, INNER);
#pragma unroll
            for (int j = 0; j < 4; ++j)
                bP[nxt][j] = load_b_frag(wbase + (size_t)(j * 16) * INNER + (i + 1) * 32, INNER);
        }
#pragma unroll
        for (int j = 0; j < 4; ++j) acc[j] = wmma_bf16(aP[cur], bP[cur][j], acc[j]);
    }

    const int colL = lane & 15, half = lane >> 4;
#pragma unroll
    for (int j = 0; j < 4; ++j) {
#pragma unroll
        for (int g = 0; g < 8; ++g) {
            int row = rt * 16 + g + 8 * half;
            int col = cg * 64 + j * 16 + colL;
            out[(size_t)row * OUP + col] = acc[j][g] + b_out[col];
        }
    }
}

// ---------------------------------------------------------------------------
// Host launcher
// ---------------------------------------------------------------------------
extern "C" void kernel_launch(void* const* d_in, const int* in_sizes, int n_in,
                              void* d_out, int out_size, void* d_ws, size_t ws_size,
                              hipStream_t stream) {
    const float* x         = (const float*)d_in[0];   // [16,1024,512]
    const float* W_qkv     = (const float*)d_in[1];   // [512,1536]
    const float* rel_table = (const float*)d_in[2];   // [3969,8]
    const float* W_out     = (const float*)d_in[3];   // [512,512]
    const float* b_out     = (const float*)d_in[4];   // [512]
    const int*   rel_index = (const int*)d_in[5];     // [1024*1024]
    float*       out       = (float*)d_out;           // [16,1024,512]

    char* ws = (char*)d_ws;
    size_t off = 0;
    auto carve = [&](size_t bytes) { char* p = ws + off; off += (bytes + 255) & ~(size_t)255; return p; };

    __bf16* xB    = (__bf16*)carve((size_t)BATCH * SEQ * INP * 2);        // 16 MB
    __bf16* WqkvT = (__bf16*)carve((size_t)QKV_COLS * INP * 2);           // 1.5 MB
    __bf16* WoutT = (__bf16*)carve((size_t)OUP * INNER * 2);              // 0.5 MB
    __bf16* qB    = (__bf16*)carve((size_t)BATCH * HEADS * SEQ * DHEAD * 2);
    __bf16* kB    = (__bf16*)carve((size_t)BATCH * HEADS * SEQ * DHEAD * 2);
    __bf16* vB    = (__bf16*)carve((size_t)BATCH * HEADS * SEQ * DHEAD * 2);
    __bf16* vtB   = (__bf16*)carve((size_t)BATCH * HEADS * DHEAD * SEQ * 2);
    __bf16* biasB = (__bf16*)carve((size_t)HEADS * SEQ * SEQ * 2);        // 16 MB
    __bf16* aoB   = (__bf16*)carve((size_t)BATCH * SEQ * INNER * 2);      // 16 MB
    (void)ws_size; (void)in_sizes; (void)n_in; (void)out_size;

    const int nX = BATCH * SEQ * INP;                 // 8388608
    cvt_to_bf16<<<(nX + 255) / 256, 256, 0, stream>>>(x, xB, nX);
    transpose_cvt<<<(INP * QKV_COLS + 255) / 256, 256, 0, stream>>>(W_qkv, WqkvT, INP, QKV_COLS);
    transpose_cvt<<<(INNER * OUP + 255) / 256, 256, 0, stream>>>(W_out, WoutT, INNER, OUP);
    const int nBias = HEADS * SEQ * SEQ;              // 8388608
    build_bias<<<(nBias + 255) / 256, 256, 0, stream>>>(rel_table, rel_index, biasB);

    qkv_gemm<<<BATCH * 64 * 24 / 8, 256, 0, stream>>>(xB, WqkvT, qB, kB, vB);
    transpose_v<<<BATCH * HEADS * 16, 256, 0, stream>>>(vB, vtB);
    attn_kernel<<<BATCH * HEADS * 8, 256, 0, stream>>>(qB, kB, vtB, biasB, aoB);
    out_gemm<<<BATCH * SEQ / 16, 256, 0, stream>>>(aoB, WoutT, b_out, out);
}
